// Convolution_22600117912286
// MI455X (gfx1250) — matrix operand order repaired and checked by reading, not compiled, not run on hardware
//
#include <hip/hip_runtime.h>
#include <hip/hip_bf16.h>

typedef __attribute__((ext_vector_type(16))) _Float16 v16h;
typedef __attribute__((ext_vector_type(8)))  float    v8f;

#define NPT   1024
#define CB    32
#define NPAD  16

__device__ __forceinline__ float fast_exp2(float x) {
#if __has_builtin(__builtin_amdgcn_exp2f)
    return __builtin_amdgcn_exp2f(x);
#else
    return exp2f(x);
#endif
}

// G[z][b][i][c] = (1/32) * sum_j W[c,i,j] * feat[z,b,j]   (i padded to 16, f16)
__global__ __launch_bounds__(256)
void conv_prep_kernel(const float* __restrict__ feat,
                      const float* __restrict__ W,
                      _Float16* __restrict__ G) {
    int t  = blockIdx.x * 256 + threadIdx.x;   // 2*1024*16*32 threads
    int c  = t & 31;
    int i  = (t >> 5) & 15;
    int zb = t >> 9;                           // z*1024 + b
    float v = 0.0f;
    if (i < 8) {
        const float* w = W + (c * 8 + i) * 8;
        const float* f = feat + zb * 8;
        float s = 0.0f;
#pragma unroll
        for (int j = 0; j < 8; ++j) s += w[j] * f[j];
        v = s * 0.03125f;                      // 1/sqrt(n_norm) = 1/32
    }
    G[t] = (_Float16)v;
}

// One 16x16 output tile per block; 8 waves each accumulate 128 b-points
// via v_wmma_f32_16x16x32_f16 (32 radial centers == K per instruction).
__global__ __launch_bounds__(256)
void conv_main_kernel(const float* __restrict__ geom,
                      const _Float16* __restrict__ G,
                      float* __restrict__ out) {
    const int tid  = threadIdx.x;
    const int lane = tid & 31;
    const int wave = tid >> 5;
    const int m    = lane & 15;     // row (A) / column (B)
    const int g    = lane >> 4;     // lane half-group
    const int tile = blockIdx.x;    // 0..127
    const int z    = tile >> 6;
    const int a0   = (tile & 63) << 4;

    const float inv_w = 31.0f / 3.5f;   // 1/width, centers c_k = k*width

    const float* pa = geom + (((z << 10) + a0 + m) * 3);
    const float ax = pa[0], ay = pa[1], az = pa[2];

    v8f acc = {};

    const int b0 = wave << 7;                                  // 128 b per wave
    const _Float16* Gb = G + (((z << 10) + b0) << 9) + (m << 5) + (g << 4);
    const float*    gb = geom + ((z << 10) + b0) * 3;
    const float ubias = (float)(g << 4);                       // +16g in K map

    for (int bb = 0; bb < 128; ++bb) {
        // uniform per wave -> scalar loads
        float bx = gb[0], by = gb[1], bz = gb[2];
        gb += 3;
        float dx = ax - bx, dy = ay - by, dz = az - bz;
        float u = __builtin_sqrtf(dx * dx + dy * dy + dz * dz + 1e-12f) * inv_w;
        float ub = u - ubias;

        v16h A;
#pragma unroll
        for (int e = 0; e < 16; ++e) {
            // K = 16g + e; basis = exp(-(u-K)^2)
            float kf = (float)e;
            float x  = ub - kf;
            A[e] = (_Float16)fast_exp2(-1.4426950408889634f * x * x);
        }

        v16h Bv = *(const v16h*)(Gb + (bb << 9));   // 32c x 16i tile column

        acc = __builtin_amdgcn_wmma_f32_16x16x32_f16(
            /*neg_a=*/false, A, /*neg_b=*/false, Bv,
            /*c_mod=*/(short)0, acc, /*reuse_a=*/false, /*reuse_b=*/false);
    }

    // cross-wave reduction of partial 16x16 f32 tiles
    __shared__ float red[8 * 256];
#pragma unroll
    for (int r = 0; r < 8; ++r) red[wave * 256 + r * 32 + lane] = acc[r];
    __syncthreads();

    float s = 0.0f;
#pragma unroll
    for (int w = 0; w < 8; ++w) s += red[w * 256 + tid];

    const int r   = tid >> 5;
    const int ls  = tid & 31;
    const int col = ls & 15;                 // output channel i
    const int row = r + ((ls >> 4) << 3);    // output point a offset
    if (col < 8)
        out[(((z << 10) + a0 + row) << 3) + col] = s;
}

extern "C" void kernel_launch(void* const* d_in, const int* in_sizes, int n_in,
                              void* d_out, int out_size, void* d_ws, size_t ws_size,
                              hipStream_t stream) {
    const float* feat = (const float*)d_in[0];   // [2,1024,8]
    const float* geom = (const float*)d_in[1];   // [2,1024,3]
    const float* W    = (const float*)d_in[2];   // [32,8,8]
    float*       out  = (float*)d_out;           // [2,1024,8]
    _Float16*    G    = (_Float16*)d_ws;         // 2*1024*16*32 halfs = 2 MB

    // G precompute: 2*1024*16*32 = 1,048,576 threads
    conv_prep_kernel<<<4096, 256, 0, stream>>>(feat, W, G);
    // main: 128 tiles (2z x 64 a-tiles), 8 waves each
    conv_main_kernel<<<128, 256, 0, stream>>>(geom, G, out);
}
